// BiMambaBlock_34119220199898
// MI455X (gfx1250) — matrix operand order repaired and checked by reading, not compiled, not run on hardware
//
#include <hip/hip_runtime.h>
#include <hip/hip_bf16.h>

// ---------------- problem constants ----------------
#define BSZ2   2
#define CCH    96
#define HHW    96
#define LSEQ   (HHW*HHW)        // 9216
#define DDI    192
#define NST    16
#define RLOW   6
#define KCONV  4
#define NDBL   (RLOW + 2*NST)   // 38
#define NDBLP  48               // padded to 3 WMMA tiles
#define MROWS  (BSZ2*LSEQ)      // 18432 rows per image
#define EPSF   1e-5f
#define NCHUNK 36
#define LCH    (LSEQ/NCHUNK)    // 256
#define NGRP   (4*2*DDI)        // 1536 (dc,b,d) state groups

typedef _Float16 v16h __attribute__((ext_vector_type(16)));
typedef _Float16 v8h  __attribute__((ext_vector_type(8)));
typedef float    v8f  __attribute__((ext_vector_type(8)));

// ---------------- WMMA fragment helpers (16x16x32 f16) ----------------
// A (16x32, row-major, lda elems): lane<16 -> K 0..7 & 16..23 ; lane>=16 -> K 8..15 & 24..31
__device__ __forceinline__ v16h load_a_frag(const _Float16* A, int lda,
                                            int mbase, int kt, int lane) {
  int row  = mbase + (lane & 15);
  int kofs = (lane & 16) ? 8 : 0;
  const _Float16* p = A + (size_t)row * lda + kt + kofs;
  v8h lo = *(const v8h*)(p);
  v8h hi = *(const v8h*)(p + 16);
  return __builtin_shufflevector(lo, hi, 0,1,2,3,4,5,6,7,8,9,10,11,12,13,14,15);
}
// B (32x16) from row-major weights W[n][k] (ldb elems): lane col = lane&15,
// lane<16 -> K 0..7 & 16..23 ; lane>=16 -> K 8..15 & 24..31
__device__ __forceinline__ v16h load_b_frag(const _Float16* __restrict__ W, int ldb,
                                            int nbase, int kt, int lane) {
  int col  = nbase + (lane & 15);
  int kofs = (lane & 16) ? 8 : 0;
  const _Float16* p = W + (size_t)col * ldb + kt + kofs;
  v8h lo = *(const v8h*)(p);
  v8h hi = *(const v8h*)(p + 16);
  return __builtin_shufflevector(lo, hi, 0,1,2,3,4,5,6,7,8,9,10,11,12,13,14,15);
}

// ---- async global->LDS stage of a contiguous tile (gfx1250 ASYNCcnt path) ----
// 16B chunks; generic LDS pointer low 32 bits == LDS byte offset (ISA aperture rule).
__device__ __forceinline__ void stage_tile_async(const _Float16* gsrc, _Float16* lds,
                                                 int nbytes) {
  unsigned ldsbase = (unsigned)(uintptr_t)lds;
  unsigned long long gbase = (unsigned long long)(uintptr_t)gsrc;
  int nchunks = nbytes >> 4;
  for (int c = threadIdx.x; c < nchunks; c += blockDim.x) {
    unsigned lofs = ldsbase + (c << 4);
    unsigned long long ga = gbase + ((unsigned long long)c << 4);
    asm volatile("global_load_async_to_lds_b128 %0, %1, off"
                 :: "v"(lofs), "v"(ga) : "memory");
  }
  asm volatile("s_wait_asynccnt 0x0" ::: "memory");
  __syncthreads();
}

// ---------------- 0: weights -> f16 (xpw zero-padded 38->48 rows) ----------------
__global__ void wprep_kernel(const float* __restrict__ inw, const float* __restrict__ xpw,
                             const float* __restrict__ outw,
                             _Float16* __restrict__ inw16, _Float16* __restrict__ xpw16,
                             _Float16* __restrict__ outw16) {
  int idx = blockIdx.x * blockDim.x + threadIdx.x;
  if (idx < 2*2*DDI*CCH) inw16[idx] = (_Float16)inw[idx];           // 2*384*96
  if (idx < 4*NDBLP*DDI) {                                          // 4*48*192
    int k = idx % DDI; int j = (idx / DDI) % NDBLP; int dc = idx / (DDI*NDBLP);
    float v = (j < NDBL) ? xpw[((size_t)dc*NDBL + j)*DDI + k] : 0.f;
    xpw16[idx] = (_Float16)v;
  }
  if (idx < 2*CCH*DDI) outw16[idx] = (_Float16)outw[idx];           // 2*96*192
}

// ---------------- 1: shuffle + LN + RMS -> short(f32), h(f16) ----------------
__global__ void prep_kernel(const float* __restrict__ x1, const float* __restrict__ x2,
                            const float* __restrict__ lnw, const float* __restrict__ lnb,
                            const float* __restrict__ rmsw,
                            float* __restrict__ shortb, _Float16* __restrict__ h16) {
  __shared__ float sm[128];
  int bid = blockIdx.x;
  int l = bid % LSEQ;
  int b = (bid / LSEQ) & 1;
  int i = bid / (2*LSEQ);
  int c = threadIdx.x;
  const float* xin = i ? x2 : x1;
  float v = 0.f;
  if (c < CCH) {
    int sc = (c & 1) * 48 + (c >> 1);                 // channel shuffle
    v = xin[((size_t)b*CCH + sc)*LSEQ + l];
  }
  sm[threadIdx.x] = (c < CCH) ? v : 0.f; __syncthreads();
  for (int s = 64; s > 0; s >>= 1) { if (threadIdx.x < s) sm[threadIdx.x] += sm[threadIdx.x+s]; __syncthreads(); }
  float mean = sm[0] * (1.f/CCH); __syncthreads();
  float dv = v - mean;
  sm[threadIdx.x] = (c < CCH) ? dv*dv : 0.f; __syncthreads();
  for (int s = 64; s > 0; s >>= 1) { if (threadIdx.x < s) sm[threadIdx.x] += sm[threadIdx.x+s]; __syncthreads(); }
  float var = sm[0] * (1.f/CCH); __syncthreads();
  int cc = (c < CCH) ? c : 0;
  float xn = dv * rsqrtf(var + EPSF) * lnw[i*CCH + cc] + lnb[i*CCH + cc];
  sm[threadIdx.x] = (c < CCH) ? xn*xn : 0.f; __syncthreads();
  for (int s = 64; s > 0; s >>= 1) { if (threadIdx.x < s) sm[threadIdx.x] += sm[threadIdx.x+s]; __syncthreads(); }
  float ms = sm[0] * (1.f/CCH);
  if (c < CCH) {
    float hv = xn * rsqrtf(ms + EPSF) * rmsw[i*CCH + c];
    size_t row = (size_t)i*MROWS + (size_t)b*LSEQ + l;
    shortb[row*CCH + c] = v;
    h16[row*CCH + c] = (_Float16)hv;
  }
}

// ---------------- 2: xz = h @ inw^T  (WMMA, A staged in LDS, 6 N-tiles/wave) ----------------
__global__ void __launch_bounds__(128)
gemm_in_kernel(const _Float16* __restrict__ h16, const _Float16* __restrict__ w16,
               float* __restrict__ xz) {
  __shared__ __align__(16) _Float16 As[16*CCH];          // 3 KB, contiguous 16-row A tile
  int i = blockIdx.z, mbase = blockIdx.x*16;
  const _Float16* A = h16 + (size_t)i * MROWS * CCH;
  const _Float16* W = w16 + (size_t)i * (2*DDI) * CCH;
  stage_tile_async(A + (size_t)mbase*CCH, As, 16*CCH*2);
  int wave = threadIdx.x >> 5, lane = threadIdx.x & 31;
  v8f acc[6] = {};
  for (int kt = 0; kt < CCH; kt += 32) {
    v16h a = load_a_frag(As, CCH, 0, kt, lane);          // ds_load from LDS
    #pragma unroll
    for (int j = 0; j < 6; ++j) {
      int nbase = (wave*6 + j)*16;
      __builtin_prefetch(W + (size_t)(nbase + (lane & 15))*CCH + kt, 0, 0);
      v16h b = load_b_frag(W, CCH, nbase, kt, lane);
      acc[j] = __builtin_amdgcn_wmma_f32_16x16x32_f16(false, a, false, b, (short)0, acc[j], false, false);
    }
  }
  float* out = xz + (size_t)i * MROWS * (2*DDI);
  int col0 = (lane & 15), rofs = (lane & 16) ? 8 : 0;
  #pragma unroll
  for (int j = 0; j < 6; ++j) {
    int col = (wave*6 + j)*16 + col0;
    for (int r = 0; r < 8; ++r)
      out[(size_t)(mbase + r + rofs)*(2*DDI) + col] = acc[j][r];
  }
}

// ---------------- 3: causal depthwise conv + SiLU -> xc(f16), both dirs ----------------
__global__ void conv_kernel(const float* __restrict__ xz, const float* __restrict__ convw,
                            const float* __restrict__ convb, _Float16* __restrict__ xc16) {
  size_t idx = (size_t)blockIdx.x * blockDim.x + threadIdx.x;
  int d  = idx % DDI;
  int t  = (idx / DDI) % LSEQ;
  int b  = (idx / ((size_t)DDI*LSEQ)) % 2;
  int dc = idx / ((size_t)DDI*LSEQ*2);     // i*2+dir
  int i  = dc >> 1, dir = dc & 1;
  const float* wr = convw + ((size_t)dc*DDI + d)*KCONV;
  float acc = convb[dc*DDI + d];
  #pragma unroll
  for (int k = 0; k < KCONV; ++k) {
    int tt = t + k - (KCONV - 1);
    float xv = 0.f;
    if (tt >= 0) {
      int lsrc = dir ? (LSEQ - 1 - tt) : tt;
      xv = xz[((size_t)i*MROWS + (size_t)b*LSEQ + lsrc)*(2*DDI) + d];
    }
    acc += wr[k] * xv;
  }
  float s = acc / (1.f + expf(-acc));
  xc16[idx] = (_Float16)s;
}

// ---------------- 4: dbl = xc @ xpw^T  (WMMA, A staged in LDS, N padded 48) ----------------
__global__ void __launch_bounds__(96)
gemm_x_kernel(const _Float16* __restrict__ xc16, const _Float16* __restrict__ w16,
              float* __restrict__ dbl) {
  __shared__ __align__(16) _Float16 As[16*DDI];          // 6 KB
  int dc = blockIdx.z, mbase = blockIdx.x*16;
  const _Float16* A = xc16 + (size_t)dc * MROWS * DDI;
  const _Float16* W = w16  + (size_t)dc * NDBLP * DDI;
  stage_tile_async(A + (size_t)mbase*DDI, As, 16*DDI*2);
  int wave = threadIdx.x >> 5, lane = threadIdx.x & 31;   // 3 waves, 1 N-tile each
  int nbase = wave*16;
  v8f c = {};
  for (int kt = 0; kt < DDI; kt += 32) {
    v16h a = load_a_frag(As, DDI, 0, kt, lane);
    v16h b = load_b_frag(W, DDI, nbase, kt, lane);
    c = __builtin_amdgcn_wmma_f32_16x16x32_f16(false, a, false, b, (short)0, c, false, false);
  }
  float* out = dbl + (size_t)dc * MROWS * NDBL;
  int col = nbase + (lane & 15), rofs = (lane & 16) ? 8 : 0;
  if (col < NDBL)
    for (int r = 0; r < 8; ++r) out[(size_t)(mbase + r + rofs)*NDBL + col] = c[r];
}

// ---------------- 5: dt = softplus(dbl[:R] @ dtw^T + dtb), once per (dc,b,t,d) ----------------
__global__ void dt_kernel(const float* __restrict__ dbl, const float* __restrict__ dtw,
                          const float* __restrict__ dtb, float* __restrict__ dtbuf) {
  size_t idx = (size_t)blockIdx.x * blockDim.x + threadIdx.x;
  int d  = idx % DDI;
  int t  = (idx / DDI) % LSEQ;
  int b  = (idx / ((size_t)DDI*LSEQ)) % 2;
  int dc = idx / ((size_t)DDI*LSEQ*2);
  const float* dr = dbl + ((size_t)dc*MROWS + (size_t)b*LSEQ + t)*NDBL;
  const float* wr = dtw + ((size_t)dc*DDI + d)*RLOW;
  float v = dtb[dc*DDI + d];
  #pragma unroll
  for (int r = 0; r < RLOW; ++r) v += dr[r] * wr[r];
  dtbuf[idx] = (v > 20.f) ? v : log1pf(expf(v));          // softplus
}

// -------- 6a: chunked scan pass 1: per-chunk local state + decay product --------
__global__ void scan1_kernel(const float* __restrict__ dbl, const _Float16* __restrict__ xc16,
                             const float* __restrict__ dtbuf, const float* __restrict__ Alog,
                             float* __restrict__ Pbuf, float* __restrict__ Hbuf) {
  int gid = blockIdx.x * blockDim.x + threadIdx.x;        // NGRP*NCHUNK*16 threads
  int n  = gid & 15;
  int g  = gid >> 4;
  int chunk = g % NCHUNK;
  int gg = g / NCHUNK;                                    // (dc*2+b)*DDI + d
  int d  = gg % DDI;
  int b  = (gg / DDI) & 1;
  int dc = gg / (DDI*2);
  float Ac = -expf(Alog[((size_t)dc*DDI + d)*NST + n]);
  int t0 = chunk * LCH;
  const float*    drow = dbl   + ((size_t)dc*MROWS + (size_t)b*LSEQ + t0)*NDBL;
  const _Float16* urow = xc16  + (((size_t)dc*2 + b)*LSEQ + t0)*(size_t)DDI;
  const float*    trow = dtbuf + (((size_t)dc*2 + b)*LSEQ + t0)*(size_t)DDI;
  float h = 0.f, P = 1.f;
  for (int tt = 0; tt < LCH; ++tt) {
    float dt = trow[(size_t)tt*DDI + d];
    float u  = (float)urow[(size_t)tt*DDI + d];
    float Bt = drow[(size_t)tt*NDBL + RLOW + n];
    float a  = expf(dt * Ac);
    h = a * h + (dt * u) * Bt;
    P *= a;
  }
  Pbuf[gid] = P;
  Hbuf[gid] = h;
}

// -------- 6b: tiny serial prefix over chunks -> per-chunk initial state --------
__global__ void scan2_kernel(const float* __restrict__ Pbuf, const float* __restrict__ Hbuf,
                             float* __restrict__ Hin) {
  int gid = blockIdx.x * blockDim.x + threadIdx.x;        // NGRP*16 threads
  int n  = gid & 15;
  int gg = gid >> 4;
  float h = 0.f;
  for (int c = 0; c < NCHUNK; ++c) {
    size_t idx = (((size_t)gg*NCHUNK + c) << 4) + n;
    Hin[idx] = h;
    h = Pbuf[idx] * h + Hbuf[idx];
  }
}

// -------- 6c: chunked scan pass 3: replay with corrected init, emit gated y --------
__global__ void scan3_kernel(const float* __restrict__ dbl, const _Float16* __restrict__ xc16,
                             const float* __restrict__ dtbuf, const float* __restrict__ xz,
                             const float* __restrict__ Alog, const float* __restrict__ Dp,
                             const float* __restrict__ Hin, float* __restrict__ ybuf) {
  int gid = blockIdx.x * blockDim.x + threadIdx.x;
  int n  = gid & 15;
  int g  = gid >> 4;
  int chunk = g % NCHUNK;
  int gg = g / NCHUNK;
  int d  = gg % DDI;
  int b  = (gg / DDI) & 1;
  int dc = gg / (DDI*2);
  int i  = dc >> 1, dir = dc & 1;
  float Ac  = -expf(Alog[((size_t)dc*DDI + d)*NST + n]);
  float Dpv = Dp[dc*DDI + d];
  int t0 = chunk * LCH;
  const float*    drow  = dbl   + ((size_t)dc*MROWS + (size_t)b*LSEQ + t0)*NDBL;
  const _Float16* urow  = xc16  + (((size_t)dc*2 + b)*LSEQ + t0)*(size_t)DDI;
  const float*    trow  = dtbuf + (((size_t)dc*2 + b)*LSEQ + t0)*(size_t)DDI;
  float*          yrow  = ybuf  + (((size_t)dc*2 + b)*LSEQ + t0)*(size_t)DDI;
  const float*    zbase = xz + ((size_t)i*MROWS + (size_t)b*LSEQ)*(2*DDI);
  float h = Hin[gid];
  for (int tt = 0; tt < LCH; ++tt) {
    float dt = trow[(size_t)tt*DDI + d];
    float u  = (float)urow[(size_t)tt*DDI + d];
    float Bt = drow[(size_t)tt*NDBL + RLOW + n];
    float Ct = drow[(size_t)tt*NDBL + RLOW + NST + n];
    float a  = expf(dt * Ac);
    h = a * h + (dt * u) * Bt;
    float yp = h * Ct;
    yp += __shfl_xor(yp, 1, 32);
    yp += __shfl_xor(yp, 2, 32);
    yp += __shfl_xor(yp, 4, 32);
    yp += __shfl_xor(yp, 8, 32);
    if (n == 0) {
      int t = t0 + tt;
      int lz = dir ? (LSEQ - 1 - t) : t;
      float z = zbase[(size_t)lz*(2*DDI) + DDI + d];
      float gate = z / (1.f + expf(-z));
      yrow[(size_t)tt*DDI + d] = (yp + Dpv * u) * gate;
    }
  }
}

// ---------------- 7: combine dirs (flip backward) -> yg(f16) ----------------
__global__ void combine_kernel(const float* __restrict__ ybuf, _Float16* __restrict__ yg16) {
  size_t idx = (size_t)blockIdx.x * blockDim.x + threadIdx.x;
  int d = idx % DDI;
  int l = (idx / DDI) % LSEQ;
  int b = (idx / ((size_t)DDI*LSEQ)) % 2;
  int i = idx / ((size_t)DDI*LSEQ*2);
  size_t fb = (((size_t)(i*2+0)*2 + b)*LSEQ + l)          * DDI + d;
  size_t bb = (((size_t)(i*2+1)*2 + b)*LSEQ + (LSEQ-1-l)) * DDI + d;
  yg16[idx] = (_Float16)(ybuf[fb] + ybuf[bb]);
}

// ---------------- 8: out = 0.5*(yg @ outw^T) + short, scatter NCHW ----------------
__global__ void __launch_bounds__(192)
gemm_out_kernel(const _Float16* __restrict__ yg16, const _Float16* __restrict__ w16,
                const float* __restrict__ shortb, float* __restrict__ out) {
  __shared__ __align__(16) _Float16 As[16*DDI];          // 6 KB
  int i = blockIdx.z, mbase = blockIdx.x*16;
  const _Float16* A = yg16 + (size_t)i * MROWS * DDI;
  const _Float16* W = w16  + (size_t)i * CCH * DDI;
  stage_tile_async(A + (size_t)mbase*DDI, As, 16*DDI*2);
  int wave = threadIdx.x >> 5, lane = threadIdx.x & 31;   // 6 waves, 1 N-tile each
  int nbase = wave*16;
  v8f c = {};
  for (int kt = 0; kt < DDI; kt += 32) {
    v16h a = load_a_frag(As, DDI, 0, kt, lane);
    v16h b = load_b_frag(W, DDI, nbase, kt, lane);
    c = __builtin_amdgcn_wmma_f32_16x16x32_f16(false, a, false, b, (short)0, c, false, false);
  }
  int col = nbase + (lane & 15), rofs = (lane & 16) ? 8 : 0;
  for (int r = 0; r < 8; ++r) {
    int m = mbase + r + rofs;
    int bb = m / LSEQ, l = m % LSEQ;
    float val = 0.5f * c[r] + shortb[((size_t)i*MROWS + m)*CCH + col];
    out[(size_t)i*(BSZ2*CCH*LSEQ) + ((size_t)bb*CCH + col)*LSEQ + l] = val;
  }
}

// ---------------- host launcher ----------------
extern "C" void kernel_launch(void* const* d_in, const int* in_sizes, int n_in,
                              void* d_out, int out_size, void* d_ws, size_t ws_size,
                              hipStream_t stream) {
  const float* x1    = (const float*)d_in[0];
  const float* x2    = (const float*)d_in[1];
  const float* inw   = (const float*)d_in[2];
  const float* convw = (const float*)d_in[3];
  const float* convb = (const float*)d_in[4];
  const float* xpw   = (const float*)d_in[5];
  const float* dtw   = (const float*)d_in[6];
  const float* dtb   = (const float*)d_in[7];
  const float* Alog  = (const float*)d_in[8];
  const float* Dp    = (const float*)d_in[9];
  const float* outw  = (const float*)d_in[10];
  const float* rmsw  = (const float*)d_in[11];
  const float* lnw   = (const float*)d_in[12];
  const float* lnb   = (const float*)d_in[13];
  float* outp = (float*)d_out;

  char* ws = (char*)d_ws;
  size_t off = 0;
  auto carve = [&](size_t bytes) -> char* {
    char* p = ws + off;
    off = (off + bytes + 255) & ~(size_t)255;
    return p;
  };
  _Float16* h16    = (_Float16*)carve((size_t)2*MROWS*CCH*2);
  float*    short32= (float*)   carve((size_t)2*MROWS*CCH*4);
  float*    xz32   = (float*)   carve((size_t)2*MROWS*2*DDI*4);
  _Float16* xc16   = (_Float16*)carve((size_t)4*MROWS*DDI*2);
  float*    dbl32  = (float*)   carve((size_t)4*MROWS*NDBL*4);
  float*    dtbuf  = (float*)   carve((size_t)4*MROWS*DDI*4);
  float*    ybuf32 = (float*)   carve((size_t)4*MROWS*DDI*4);
  _Float16* yg16   = (_Float16*)carve((size_t)2*MROWS*DDI*2);
  float*    Pbuf   = (float*)   carve((size_t)NGRP*NCHUNK*NST*4);
  float*    Hbuf   = (float*)   carve((size_t)NGRP*NCHUNK*NST*4);
  float*    Hin    = (float*)   carve((size_t)NGRP*NCHUNK*NST*4);
  _Float16* inw16  = (_Float16*)carve((size_t)2*2*DDI*CCH*2);
  _Float16* xpw16  = (_Float16*)carve((size_t)4*NDBLP*DDI*2);
  _Float16* outw16 = (_Float16*)carve((size_t)2*CCH*DDI*2);

  wprep_kernel<<<288, 256, 0, stream>>>(inw, xpw, outw, inw16, xpw16, outw16);
  prep_kernel<<<2*BSZ2*LSEQ, 128, 0, stream>>>(x1, x2, lnw, lnb, rmsw, short32, h16);
  gemm_in_kernel<<<dim3(MROWS/16, 1, 2), 128, 0, stream>>>(h16, inw16, xz32);
  conv_kernel<<<(4*2*LSEQ*DDI)/256, 256, 0, stream>>>(xz32, convw, convb, xc16);
  gemm_x_kernel<<<dim3(MROWS/16, 1, 4), 96, 0, stream>>>(xc16, xpw16, dbl32);
  dt_kernel<<<(4*2*LSEQ*DDI)/256, 256, 0, stream>>>(dbl32, dtw, dtb, dtbuf);
  scan1_kernel<<<(NGRP*NCHUNK*NST)/256, 256, 0, stream>>>(dbl32, xc16, dtbuf, Alog, Pbuf, Hbuf);
  scan2_kernel<<<(NGRP*NST)/256, 256, 0, stream>>>(Pbuf, Hbuf, Hin);
  scan3_kernel<<<(NGRP*NCHUNK*NST)/256, 256, 0, stream>>>(dbl32, xc16, dtbuf, xz32, Alog, Dp, Hin, ybuf32);
  combine_kernel<<<(2*BSZ2*LSEQ*DDI)/256, 256, 0, stream>>>(ybuf32, yg16);
  gemm_out_kernel<<<dim3(MROWS/16, 1, 2), 192, 0, stream>>>(yg16, outw16, short32, outp);
}
